// BasicBlockBit_3839700763105
// MI455X (gfx1250) — matrix operand order repaired and checked by reading, not compiled, not run on hardware
//
#include <hip/hip_runtime.h>

// ---------------------------------------------------------------------------
// BitNet BasicBlock for gfx1250: ternary-quantized 3x3 convs as implicit GEMM
// using v_wmma_f32_16x16x32_f16 (wave32 WMMA), fused BN/SiLU/residual.
// Tiling v4: 128 oc x 128 pixels (16w x 8h) per block; each A fragment feeds
// 8 WMMAs; B fragments double-buffered so ds_load latency hides behind the
// previous WMMA instead of a full s_wait_dscnt 0 before every WMMA.
// ---------------------------------------------------------------------------

typedef __attribute__((ext_vector_type(16))) _Float16 v16h;
typedef __attribute__((ext_vector_type(8)))  _Float16 v8h;
typedef __attribute__((ext_vector_type(8)))  float    v8f;

#define C_IN    128
#define HW      112
#define NBATCH  32
#define NTAPS   9
#define KTOT    (NTAPS * C_IN)        // 1152
#define NCHUNK  (KTOT / 32)           // 36 (K=32 per f16 WMMA)
#define WELEMS  (C_IN * C_IN * NTAPS) // 147456
#define LDS_PIX_STRIDE 136            // 128 ch + 8 pad f16 -> conflict-free B loads
#define ROWS_PB 8                     // output rows per block (112 = 14*8)
#define HBLK    (HW / ROWS_PB)        // 14
#define HALO_PIX ((ROWS_PB + 2) * 18) // 180 staged pixels

__device__ __forceinline__ float silu_f(float x) {
    return x / (1.0f + __expf(-x));
}

// ---------------------------------------------------------------------------
// Kernel 1: approximate per-tensor median of |w| via 4096-bin LDS histogram.
// ---------------------------------------------------------------------------
__global__ __launch_bounds__(256)
void median_scale_kernel(const float* __restrict__ w1,
                         const float* __restrict__ w2,
                         float* __restrict__ scales) {
    __shared__ unsigned hist[4096];
    const float* w = (blockIdx.x == 0) ? w1 : w2;
    for (int i = threadIdx.x; i < 4096; i += 256) hist[i] = 0u;
    __syncthreads();
    for (int i = threadIdx.x; i < WELEMS; i += 256) {
        float a = fabsf(w[i]);                // |w| ~ 0.05*|N(0,1)|, << 1
        int bin = (int)(a * 4096.0f);
        if (bin > 4095) bin = 4095;
        atomicAdd(&hist[bin], 1u);
    }
    __syncthreads();
    if (threadIdx.x == 0) {
        const unsigned target = WELEMS / 2;
        unsigned cum = 0; int b = 0;
        for (; b < 4096; ++b) { cum += hist[b]; if (cum >= target) break; }
        float s = ((float)b + 0.5f) * (1.0f / 4096.0f);
        scales[blockIdx.x] = fmaxf(s, 1e-8f);
    }
}

// ---------------------------------------------------------------------------
// Kernel 2: fold weight-scale + bias + BN into per-channel fma constants:
//   y = conv_ternary * (wscale*inv) + (bias*inv + beta - mean*inv)
// ---------------------------------------------------------------------------
__global__ __launch_bounds__(128)
void bn_prep_kernel(const float* __restrict__ scaleP,
                    const float* __restrict__ bias,
                    const float* __restrict__ g,  const float* __restrict__ be,
                    const float* __restrict__ m,  const float* __restrict__ v,
                    float* __restrict__ fa, float* __restrict__ fb) {
    const int c = threadIdx.x;
    const float inv = g[c] * rsqrtf(v[c] + 1e-5f);
    fa[c] = scaleP[0] * inv;
    fb[c] = bias[c] * inv + be[c] - m[c] * inv;
}

// ---------------------------------------------------------------------------
// Kernel 3: ternary quantize weights -> f16 in the WMMA A-fragment layout:
// wq[(chunk*128 + oc)*32 + slot], K = (kh*3+kw)*128 + c, chunk = K/32.
// Slot = K%32 (contiguous k order, same as the B loads; lane-half 0 reads
// K 0..15, half 1 reads K 16..31) so each lane's 16 f16 A
// elements are one contiguous 32B run.
// ---------------------------------------------------------------------------
__global__ __launch_bounds__(256)
void quantize_kernel(const float* __restrict__ w,      // [128][128][3][3]
                     const float* __restrict__ scaleP,
                     _Float16* __restrict__ wq) {      // [36][128][32]
    const int i = blockIdx.x * 256 + threadIdx.x;
    if (i >= WELEMS) return;
    const float inv = 1.0f / scaleP[0];
    const int t   = i % NTAPS;           // kh*3+kw
    const int rem = i / NTAPS;
    const int c   = rem % C_IN;
    const int oc  = rem / C_IN;
    float q = rintf(w[i] * inv);         // RNE == jnp.round
    q = fminf(1.0f, fmaxf(-1.0f, q));    // ternary {-1,0,1}; scale folded into fa
    const int K     = t * C_IN + c;
    const int chunk = K >> 5;
    const int kk    = K & 31;
    const int slot  = kk;
    wq[((size_t)chunk * C_IN + oc) * 32 + slot] = (_Float16)q;
}

// ---------------------------------------------------------------------------
// Kernel 4: NCHW fp32 -> NHWC f16, LDS-transposed so both global sides coalesce.
// ---------------------------------------------------------------------------
__global__ __launch_bounds__(256)
void nchw_to_nhwc_f16_kernel(const float* __restrict__ x,
                             _Float16* __restrict__ dst) {
    __shared__ float sT[HW * 129];                 // [w][c], pad -> conflict-free
    const int h = blockIdx.x % HW;
    const int n = blockIdx.x / HW;
    const size_t base = (size_t)n * C_IN * HW * HW + (size_t)h * HW;
    for (int i = threadIdx.x; i < C_IN * HW; i += 256) {
        const int c = i / HW;
        const int w = i - c * HW;                  // lanes sweep w -> coalesced read
        sT[w * 129 + c] = x[base + (size_t)c * HW * HW + w];
    }
    __syncthreads();
    _Float16* out = dst + (size_t)(n * HW + h) * HW * C_IN;
    for (int i = threadIdx.x; i < HW * C_IN; i += 256) {
        const int w = i >> 7;
        const int c = i & 127;                     // lanes sweep c -> coalesced write
        out[(size_t)w * C_IN + c] = (_Float16)sT[w * 129 + c];
    }
}

// ---------------------------------------------------------------------------
// Kernel 5: 3x3 conv, implicit GEMM via WMMA.
//   Block = 256 threads = 8 waves. Tile = 128 oc x 128 pixels (16 w x 8 rows).
//   Wave w owns oc [16w, 16w+16) and all 8 pixel rows (8 accumulators):
//   one A-fragment load -> 8 WMMAs; B double-buffered one row ahead.
//   K-loop: 9 taps (rolled) x 4 channel-chunks x 8 rows = 288 WMMA / wave.
//   STAGE 1: out = f16 NHWC silu(bn1(conv1+b1))    -> workspace
//   STAGE 2: out = fp32 NCHW silu(bn2(conv2+b2)+x) -> d_out
// ---------------------------------------------------------------------------
struct U32x8 { uint4 lo, hi; };   // 32B, 16B-aligned carrier for v16h LDS loads

template<int STAGE>
__global__ __launch_bounds__(256, 4)   // cap <=256 VGPRs, >=4 waves/SIMD
void conv3x3_wmma_kernel(const _Float16* __restrict__ src,    // NHWC f16 activations
                         const _Float16* __restrict__ wq,     // [36][128][32] A-frags
                         const float* __restrict__ fa,        // per-oc fma scale
                         const float* __restrict__ fb,        // per-oc fma shift
                         const float* __restrict__ identity,  // NCHW fp32 (STAGE 2)
                         _Float16* __restrict__ dstH,         // NHWC f16  (STAGE 1)
                         float* __restrict__ dstF)            // NCHW fp32 (STAGE 2)
{
    // Halo patch: 10 rows x 18 cols x 128 ch, pixel stride padded to 136 f16.
    __shared__ __align__(16) _Float16 sAct[(ROWS_PB + 2) * 18 * LDS_PIX_STRIDE];

    const int bx   = blockIdx.x;
    const int wseg = bx % 7;                  // 112 = 7 * 16
    const int hb   = (bx / 7) % HBLK;
    const int n    = bx / (7 * HBLK);
    const int h0   = hb * ROWS_PB;
    const int w0   = wseg * 16;
    const int tid  = threadIdx.x;

    // ---- Stage halo patch: 180 pixels x 256B of channels, 16B per thread-slot
    for (int i = tid; i < HALO_PIX * 16; i += 256) {
        const int p  = i >> 4;                // pixel (row*18+col)
        const int v  = i & 15;                // 16B sub-vector of 256B channel run
        const int r  = p / 18;
        const int cc = p - r * 18;
        const int gh = h0 + r - 1;
        const int gw = w0 + cc - 1;
        uint4 val = make_uint4(0u, 0u, 0u, 0u);
        if ((unsigned)gh < HW && (unsigned)gw < HW) {
            val = *(const uint4*)(src + (((size_t)(n * HW + gh) * HW + gw) * C_IN + v * 8));
        }
        *(uint4*)(&sAct[p * LDS_PIX_STRIDE + v * 8]) = val;
    }
    __syncthreads();

    const int wave = tid >> 5;
    const int lane = tid & 31;
    const int row  = lane & 15;               // A row (oc) / B col (pixel)
    const int half = lane >> 4;               // K half-select
    const int ocb  = wave * 16;

    v8f acc[ROWS_PB];
    #pragma unroll
    for (int j = 0; j < ROWS_PB; ++j)
        acc[j] = (v8f){0.f, 0.f, 0.f, 0.f, 0.f, 0.f, 0.f, 0.f};

    const _Float16* aptr = wq + ((size_t)(ocb + row)) * 32 + half * 16;

    #pragma unroll 1                          // keep VGPR working set bounded
    for (int tap = 0; tap < NTAPS; ++tap) {
        const int dh = tap / 3;               // 0..2
        const int dw = tap - dh * 3;          // 0..2
        const _Float16* ap = aptr + (size_t)tap * 4 * C_IN * 32;
        const int bbase = (dh * 18 + row + dw) * LDS_PIX_STRIDE + half * 16;

        #pragma unroll
        for (int cb = 0; cb < 4; ++cb) {      // 4 channel chunks of 32
            // One A fragment (32B contiguous per lane), reused for 8 pixel rows.
            v16h a = *(const v16h*)(ap + (size_t)cb * C_IN * 32);

            // B double-buffer: prefetch row j+1 before the WMMA consuming row j,
            // so the s_wait before each WMMA covers an already-in-flight load.
            U32x8 cur = *(const U32x8*)(&sAct[bbase + cb * 32]);
            #pragma unroll
            for (int j = 0; j < ROWS_PB; ++j) {
                U32x8 nxt;
                if (j + 1 < ROWS_PB)
                    nxt = *(const U32x8*)(&sAct[bbase + (j + 1) * (18 * LDS_PIX_STRIDE) + cb * 32]);
                v16h b = __builtin_bit_cast(v16h, cur);
                acc[j] = __builtin_amdgcn_wmma_f32_16x16x32_f16(
                    false, a, false, b, (short)0, acc[j], false, false);
                if (j + 1 < ROWS_PB) cur = nxt;
            }
        }
    }

    // ---- Epilogue: D layout -> lane holds pixel = lane&15, oc = ocb+8*half+v
    const int pix = row;
    const int ocs = ocb + half * 8;

    float fav[8], fbv[8];
    #pragma unroll
    for (int v = 0; v < 8; ++v) { fav[v] = fa[ocs + v]; fbv[v] = fb[ocs + v]; }

    #pragma unroll
    for (int j = 0; j < ROWS_PB; ++j) {
        const int h = h0 + j;
        if (STAGE == 1) {
            v8h ov;
            #pragma unroll
            for (int v = 0; v < 8; ++v)
                ov[v] = (_Float16)silu_f(acc[j][v] * fav[v] + fbv[v]);
            *(v8h*)(dstH + (((size_t)(n * HW + h) * HW + (w0 + pix)) * C_IN + ocs)) = ov;
        } else {
            #pragma unroll
            for (int v = 0; v < 8; ++v) {
                const size_t idx = (((size_t)n * C_IN + (ocs + v)) * HW + h) * HW + (w0 + pix);
                float y = acc[j][v] * fav[v] + fbv[v] + identity[idx];
                dstF[idx] = silu_f(y);
            }
        }
    }
}

// ---------------------------------------------------------------------------
// Host launcher
// ---------------------------------------------------------------------------
extern "C" void kernel_launch(void* const* d_in, const int* in_sizes, int n_in,
                              void* d_out, int out_size, void* d_ws, size_t ws_size,
                              hipStream_t stream) {
    const float* x   = (const float*)d_in[0];
    const float* w1  = (const float*)d_in[1];
    const float* b1  = (const float*)d_in[2];
    const float* g1  = (const float*)d_in[3];
    const float* be1 = (const float*)d_in[4];
    const float* m1  = (const float*)d_in[5];
    const float* v1  = (const float*)d_in[6];
    const float* w2  = (const float*)d_in[7];
    const float* b2  = (const float*)d_in[8];
    const float* g2  = (const float*)d_in[9];
    const float* be2 = (const float*)d_in[10];
    const float* m2  = (const float*)d_in[11];
    const float* v2  = (const float*)d_in[12];
    float* out = (float*)d_out;

    // Workspace layout (256B-aligned regions):
    //   +0    : scales[2]
    //   +256  : fa1,fb1,fa2,fb2  (4 * 128 f32 = 2048 B)
    //   +4096 : wq1 (36*128*32 f16 = 294,912 B), then wq2
    //   then  : x as NHWC f16 (102.8 MB), then conv1 output NHWC f16
    char* wsb = (char*)d_ws;
    float* scales = (float*)wsb;
    float* fa1 = (float*)(wsb + 256);
    float* fb1 = fa1 + C_IN;
    float* fa2 = fb1 + C_IN;
    float* fb2 = fa2 + C_IN;
    const size_t WQ_BYTES  = (size_t)NCHUNK * C_IN * 32 * sizeof(_Float16);
    const size_t ACT_ELEMS = (size_t)NBATCH * HW * HW * C_IN;
    _Float16* wq1 = (_Float16*)(wsb + 4096);
    _Float16* wq2 = (_Float16*)(wsb + 4096 + WQ_BYTES);
    _Float16* xh  = (_Float16*)(wsb + 4096 + 2 * WQ_BYTES);
    _Float16* a1  = xh + ACT_ELEMS;

    median_scale_kernel<<<2, 256, 0, stream>>>(w1, w2, scales);
    bn_prep_kernel<<<1, 128, 0, stream>>>(scales + 0, b1, g1, be1, m1, v1, fa1, fb1);
    bn_prep_kernel<<<1, 128, 0, stream>>>(scales + 1, b2, g2, be2, m2, v2, fa2, fb2);
    quantize_kernel<<<WELEMS / 256, 256, 0, stream>>>(w1, scales + 0, wq1);
    quantize_kernel<<<WELEMS / 256, 256, 0, stream>>>(w2, scales + 1, wq2);
    nchw_to_nhwc_f16_kernel<<<NBATCH * HW, 256, 0, stream>>>(x, xh);

    const int blocks = NBATCH * HBLK * 7;   // 3136: (n, 8-row band, 16-wide w segment)
    conv3x3_wmma_kernel<1><<<blocks, 256, 0, stream>>>(
        xh, wq1, fa1, fb1, nullptr, a1, nullptr);
    conv3x3_wmma_kernel<2><<<blocks, 256, 0, stream>>>(
        a1, wq2, fa2, fb2, x, nullptr, out);
}
